// ISTFT_25855703122688
// MI455X (gfx1250) — compile-verified, matrix operand order checked by reading, not run on hardware
//
#include <hip/hip_runtime.h>
#include <math.h>

// ---------------- types for WMMA ----------------
typedef __bf16 bf16_t;
typedef bf16_t v16bf __attribute__((ext_vector_type(16)));
typedef bf16_t v8bf  __attribute__((ext_vector_type(8)));
typedef float  v8f   __attribute__((ext_vector_type(8)));

// ---------------- problem constants ----------------
#define NFFT      512
#define HOP       128
#define NFREQ     257               // onesided bins
#define TFRAMES   2048
#define BC        32                // B*C = 16*2
#define K2        (2 * NFREQ)       // 514: interleaved re/im along K
#define KPAD      544               // 17 chunks of 32
#define NCHUNK    17
#define OUTLEN    ((TFRAMES - 1) * HOP)        // 262016 per (b,c)
#define FULLLEN   (NFFT + (TFRAMES - 1) * HOP) // 262528
#define XBCSTRIDE (NFREQ * TFRAMES * 2)        // floats per (b,c)
#define TTILE     32                           // frames per block
#define TILES_PER_BC (TFRAMES / TTILE)         // 64
#define REGION    (NFFT + (TTILE - 1) * HOP)   // 4480 samples per block
#define PAD       (NFFT / 2)                   // 256 trim
#define TWO_PI_OVER_N 0.012271846303085129f    // 2*pi/512

// =====================================================================
// Kernel 0: build windowed irfft basis A[n][k2] (bf16, zero-padded K)
//   x[n] = (1/512) [ Xr0 + 2*sum_{k=1..255}(Xr cos - Xi sin) + Xr256 cos(pi n) ]
//   window (periodic hann) folded in; imag coeff at k=0,256 zeroed.
// =====================================================================
__global__ __launch_bounds__(256) void build_basis(bf16_t* __restrict__ A) {
    int idx = blockIdx.x * 256 + threadIdx.x;
    if (idx >= NFFT * KPAD) return;
    int n  = idx / KPAD;
    int k2 = idx - n * KPAD;
    int k  = k2 >> 1;
    int ri = k2 & 1;
    float v = 0.0f;
    if (k <= 256 && !(ri == 1 && (k == 0 || k == 256))) {
        float scale = (k == 0 || k == 256) ? (1.0f / 512.0f) : (2.0f / 512.0f);
        int m = (k * n) & (NFFT - 1);          // exact angle reduction (period 512)
        float ang = TWO_PI_OVER_N * (float)m;
        float trig = (ri == 0) ? cosf(ang) : -sinf(ang);
        float w = 0.5f - 0.5f * cosf(TWO_PI_OVER_N * (float)n);
        v = scale * w * trig;
    }
    A[idx] = (bf16_t)v;
}

// =====================================================================
// Kernel 1: zero the output (it is the overlap-add accumulator)
// =====================================================================
__global__ __launch_bounds__(256) void zero_out(float* __restrict__ out, int n) {
    for (int i = blockIdx.x * blockDim.x + threadIdx.x; i < n;
         i += gridDim.x * blockDim.x)
        out[i] = 0.0f;
}

// =====================================================================
// Kernel 2: fused GEMM (bf16 WMMA) + windowed overlap-add
//   block = (bc, 32-frame tile): frames[512][32] = A[512xK] * X[Kx32]
//   8 waves; wave w owns m-tiles 4w..4w+3 and both 16-col n-tiles.
// =====================================================================
__global__ __launch_bounds__(256)
void istft_gemm(const float* __restrict__ X, const bf16_t* __restrict__ A,
                float* __restrict__ out) {
    __shared__ __align__(16) bf16_t sB[TTILE][40];   // [t][k2] chunk, 16B-aligned rows
    __shared__ float sOut[REGION];                   // local OLA accumulator

    const int tid  = threadIdx.x;
    const int lane = tid & 31;
    const int wave = tid >> 5;
    const int tile = blockIdx.x & (TILES_PER_BC - 1);
    const int bc   = blockIdx.x >> 6;
    const int t0   = tile * TTILE;
    const float* Xbc = X + (size_t)bc * XBCSTRIDE;

    for (int i = tid; i < REGION; i += 256) sOut[i] = 0.0f;

    v8f acc[4][2];
#pragma unroll
    for (int mi = 0; mi < 4; ++mi)
#pragma unroll
        for (int ni = 0; ni < 2; ++ni)
            acc[mi][ni] = (v8f){0.f, 0.f, 0.f, 0.f, 0.f, 0.f, 0.f, 0.f};

    // staging assignment: thread -> (t = tid&31, kpair = tid>>5 and +8)
    const int t_st = tid & 31;
    const int kk   = tid >> 5;

    for (int c = 0; c < NCHUNK; ++c) {
        __syncthreads();   // sB reads from previous chunk done
        // ---- stage X chunk (32 k2 x 32 t) into LDS as bf16 [t][k2] ----
#pragma unroll
        for (int h = 0; h < 2; ++h) {
            int kp = kk + 8 * h;            // 0..15 within chunk (complex bins)
            int k  = c * 16 + kp;
            float re = 0.0f, im = 0.0f;
            if (k <= 256) {
                const float* p =
                    Xbc + (size_t)k * (TFRAMES * 2) + (size_t)(t0 + t_st) * 2;
                re = p[0];
                im = p[1];
            }
            sB[t_st][2 * kp]     = (bf16_t)re;
            sB[t_st][2 * kp + 1] = (bf16_t)im;
        }
        __syncthreads();

        // ---- B fragments (32x16 bf16 per n-tile), from LDS ----
        v16bf bfrag[2];
#pragma unroll
        for (int ni = 0; ni < 2; ++ni) {
            const bf16_t* bp = &sB[(lane & 15) + 16 * ni][(lane >> 4) * 8];
            v8bf lo = *(const v8bf*)bp;
            v8bf hi = *(const v8bf*)(bp + 16);
            bfrag[ni] = __builtin_shufflevector(lo, hi, 0, 1, 2, 3, 4, 5, 6, 7,
                                                8, 9, 10, 11, 12, 13, 14, 15);
        }

        // ---- A fragments from global (L2-resident basis) + WMMA ----
#pragma unroll
        for (int mi = 0; mi < 4; ++mi) {
            int n = (wave * 4 + mi) * 16 + (lane & 15);
            const bf16_t* ap = A + (size_t)n * KPAD + c * 32 + (lane >> 4) * 8;
            v8bf lo = *(const v8bf*)ap;
            v8bf hi = *(const v8bf*)(ap + 16);
            v16bf afrag = __builtin_shufflevector(
                lo, hi, 0, 1, 2, 3, 4, 5, 6, 7, 8, 9, 10, 11, 12, 13, 14, 15);
            acc[mi][0] = __builtin_amdgcn_wmma_f32_16x16x32_bf16(
                false, afrag, false, bfrag[0], (short)0, acc[mi][0], false, false);
            acc[mi][1] = __builtin_amdgcn_wmma_f32_16x16x32_bf16(
                false, afrag, false, bfrag[1], (short)0, acc[mi][1], false, false);
        }
    }
    __syncthreads();

    // ---- overlap-add accumulators into LDS region (p = 128*t + n) ----
#pragma unroll
    for (int mi = 0; mi < 4; ++mi) {
        int nbase = (wave * 4 + mi) * 16 + 8 * (lane >> 4);  // C layout: M=r+8*hi
#pragma unroll
        for (int ni = 0; ni < 2; ++ni) {
            int tl = ni * 16 + (lane & 15);                  // C layout: N=lane&15
#pragma unroll
            for (int r = 0; r < 8; ++r)
                atomicAdd(&sOut[tl * HOP + nbase + r], acc[mi][ni][r]);
        }
    }
    __syncthreads();

    // ---- region -> global (edges overlap neighbor blocks), trim center pad ----
    float* obc = out + (size_t)bc * OUTLEN;
    const int pbase = t0 * HOP;
    for (int i = tid; i < REGION; i += 256) {
        int o = pbase + i - PAD;
        if (o >= 0 && o < OUTLEN) atomicAdd(&obc[o], sOut[i]);
    }
}

// =====================================================================
// Kernel 3: divide by window-envelope (<=4 hann^2 terms per sample)
// =====================================================================
__global__ __launch_bounds__(256) void normalize(float* __restrict__ out, int n) {
    for (int idx = blockIdx.x * blockDim.x + threadIdx.x; idx < n;
         idx += gridDim.x * blockDim.x) {
        int ol = idx % OUTLEN;
        int p  = ol + PAD;
        int tmax = p >> 7;
        if (tmax > TFRAMES - 1) tmax = TFRAMES - 1;
        int a = p - (NFFT - 1);
        int tmin = (a > 0) ? ((a + HOP - 1) >> 7) : 0;
        float env = 0.0f;
        for (int t = tmin; t <= tmax; ++t) {
            int nn = p - t * HOP;
            float w = 0.5f - 0.5f * cosf(TWO_PI_OVER_N * (float)nn);
            env += w * w;
        }
        out[idx] = out[idx] / env;
    }
}

// =====================================================================
extern "C" void kernel_launch(void* const* d_in, const int* in_sizes, int n_in,
                              void* d_out, int out_size, void* d_ws, size_t ws_size,
                              hipStream_t stream) {
    const float* X = (const float*)d_in[0];  // (16,2,257,2048,2) f32
    // d_in[1] is the hann window; identical analytic form is folded into A.
    float* out = (float*)d_out;              // (16,2,262016) f32
    bf16_t* A  = (bf16_t*)d_ws;              // 512*544*2 B = 557 KB scratch

    build_basis<<<(NFFT * KPAD + 255) / 256, 256, 0, stream>>>(A);
    zero_out<<<2048, 256, 0, stream>>>(out, out_size);
    istft_gemm<<<BC * TILES_PER_BC, 256, 0, stream>>>(X, A, out);
    normalize<<<2048, 256, 0, stream>>>(out, out_size);
}
